// Signed_GAT_16982300688849
// MI455X (gfx1250) — matrix operand order repaired and checked by reading, not compiled, not run on hardware
//
#include <hip/hip_runtime.h>

// ---------------------------------------------------------------------------
// Signed GAT on MI455X (gfx1250, wave32).
//   e[i,j]      = lrelu(f_i + g_j)  with f = X@(Wd@a1), g = X@(Wd@a2)  (rank-1)
//   S>1/N mask  = e[i,j] > t_i,  t_i = rowmax + ln(rowsum) - ln(N), packed bits
//   att@X       = masked-softmax aggregation via V_WMMA_F32_16X16X32_BF16
//                 (A = bf16 weight tile from double-buffered LDS,
//                  B = pre-transposed bf16 X resident in L2, C/D = f32)
// ---------------------------------------------------------------------------

typedef __attribute__((ext_vector_type(16))) __bf16 v16bf;
typedef __attribute__((ext_vector_type(8)))  __bf16 v8bf;
typedef __attribute__((ext_vector_type(8)))  float  v8f;

#define NV    4096
#define DM    300
#define PITCH 320          // padded feature dim (20 tiles of 16)
#define ALPHA 0.3f

__device__ __forceinline__ float lrelu(float x) { return fmaxf(x, ALPHA * x); }

__device__ __forceinline__ float wave_sum(float v) {
    for (int o = 16; o; o >>= 1) v += __shfl_xor(v, o, 32);
    return v;
}
__device__ __forceinline__ float wave_max(float v) {
    for (int o = 16; o; o >>= 1) v = fmaxf(v, __shfl_xor(v, o, 32));
    return v;
}
__device__ __forceinline__ float wave_min(float v) {
    for (int o = 16; o; o >>= 1) v = fminf(v, __shfl_xor(v, o, 32));
    return v;
}

// ---- K0: six 300-vectors  M @ a  (Wd@a1, Wd@a2, W1@ah_lo, W1@ah_hi, Wo@ao_lo, Wo@ao_hi)
__global__ void k_smallvec(const float* Wd, const float* a1, const float* a2,
                           const float* W1, const float* ah,
                           const float* Wo, const float* ao, float* vecs) {
    int t = blockIdx.x * blockDim.x + threadIdx.x;
    if (t >= 6 * DM) return;
    int v = t / DM, r = t % DM;
    const float* M;
    const float* a;
    if (v < 2)      { M = Wd; a = (v == 0) ? a1 : a2; }
    else if (v < 4) { M = W1; a = ah + ((v == 3) ? DM : 0); }
    else            { M = Wo; a = ao + ((v == 5) ? DM : 0); }
    float s = 0.f;
    for (int k = 0; k < DM; ++k) s += M[r * DM + k] * a[k];
    vecs[v * 304 + r] = s;
}

// ---- projections: f[i] = X[i,:]·va, g[i] = X[i,:]·vb   (one wave per row)
__global__ void k_proj2(const float* X, int pitch, const float* va, const float* vb,
                        float* f, float* g) {
    int lane = threadIdx.x & 31, wid = threadIdx.x >> 5;
    int row = blockIdx.x * (blockDim.x >> 5) + wid;
    if (row >= NV) return;
    const float* xr = X + (size_t)row * pitch;
    float s0 = 0.f, s1 = 0.f;
    for (int d = lane; d < DM; d += 32) {
        float x = xr[d];
        s0 += x * va[d];
        s1 += x * vb[d];
    }
    s0 = wave_sum(s0);
    s1 = wave_sum(s1);
    if (lane == 0) { f[row] = s0; g[row] = s1; }
}

// ---- transpose + f32->bf16:  XT[d][i] = bf16(X[i][d])  (zero pad d>=300)
__global__ void k_transpose_bf(const float* X, int pitch, __bf16* XT) {
    int i = blockIdx.x * blockDim.x + threadIdx.x;   // 0..4095
    int d = blockIdx.y;                              // 0..319
    float v = (d < DM) ? X[(size_t)i * pitch + d] : 0.f;
    XT[(size_t)d * NV + i] = (__bf16)v;
}

// ---- S row threshold: t_i = max_j lrelu(f_i+g_j) + ln(sum exp(.-max)) - ln(N)
__global__ void k_sthresh(const float* f, const float* g, float* thr) {
    int lane = threadIdx.x & 31, wid = threadIdx.x >> 5;
    int row = blockIdx.x * (blockDim.x >> 5) + wid;
    if (row >= NV) return;
    float fi = f[row];
    float m = -__builtin_inff();
    for (int j = lane; j < NV; j += 32) m = fmaxf(m, lrelu(fi + g[j]));
    m = wave_max(m);
    float z = 0.f;
    for (int j = lane; j < NV; j += 32) z += __expf(lrelu(fi + g[j]) - m);
    z = wave_sum(z);
    if (lane == 0) thr[row] = m + __logf(z) - __logf((float)NV);
}

// ---- mask build (optional) + masked softmax stats for scores lrelu(p_i+q_j)
// stats layout: [0]=m_pos, [1]=1/Z_pos, [2]=m_neg, [3]=1/Z_neg  (each stride NV)
__global__ void k_mask_stats(const float* f, const float* g, const float* thr,
                             const int* adj, const float* p, const float* q,
                             unsigned* mask, float* stats, int build) {
    int lane = threadIdx.x & 31, wid = threadIdx.x >> 5;
    int row = blockIdx.x * (blockDim.x >> 5) + wid;
    if (row >= NV) return;
    float fi = build ? f[row] : 0.f;
    float tr = build ? thr[row] : 0.f;
    float pi = p[row];
    unsigned own[4] = {0u, 0u, 0u, 0u};
    float smax = -__builtin_inff(), smin = __builtin_inff();
    for (int c = 0; c < 128; ++c) {
        int j = c * 32 + lane;
        bool bit;
        if (build) {
            float e = lrelu(fi + g[j]);
            bit = (e > tr) || (adj[(size_t)row * NV + j] != 0);
            unsigned w = (unsigned)__ballot(bit);
            if (lane == 0) mask[(size_t)row * 128 + c] = w;
        } else {
            unsigned w = mask[(size_t)row * 128 + c];
            bit = (w >> lane) & 1u;
        }
        if (bit) {
            float s = lrelu(pi + q[j]);
            smax = fmaxf(smax, s);
            smin = fminf(smin, s);
            own[c >> 5] |= 1u << (c & 31);
        }
    }
    smax = wave_max(smax);
    smin = wave_min(smin);
    float mp = smax, mn = -smin;
    float zp = 0.f, zn = 0.f;
    for (int c = 0; c < 128; ++c) {
        if ((own[c >> 5] >> (c & 31)) & 1u) {
            float s = lrelu(pi + q[c * 32 + lane]);
            zp += __expf(s - mp);
            zn += __expf(-s - mn);
        }
    }
    zp = wave_sum(zp);
    zn = wave_sum(zn);
    if (lane == 0) {
        stats[0 * NV + row] = mp;
        stats[1 * NV + row] = (zp > 0.f) ? 1.f / zp : 0.f;
        stats[2 * NV + row] = mn;
        stats[3 * NV + row] = (zn > 0.f) ? 1.f / zn : 0.f;
    }
}

// ---- heavy kernel: aggP = softmax_pos @ X, aggN = softmax_neg @ X  (bf16 WMMA)
// one block = 16 output rows; 8 waves: even waves -> pos, odd waves -> neg,
// each wave owns 5 of the 20 column tiles. Weight tiles are double-buffered
// in LDS (one barrier per k-chunk); weight math is branchless VALU/TRANS that
// co-executes with the WMMAs.
__global__ __launch_bounds__(256) void k_agg_wmma(const __bf16* __restrict__ XT,
                                                  const float* __restrict__ p,
                                                  const float* __restrict__ q,
                                                  const float* __restrict__ stats,
                                                  const unsigned* __restrict__ mask,
                                                  float* __restrict__ aggP,
                                                  float* __restrict__ aggN) {
    __shared__ __align__(16) __bf16 Wlds[2][2][16 * 32];   // [buf][sign][m*32+k]

    const int tid = threadIdx.x;
    const int rowbase = blockIdx.x * 16;
    const int lane = tid & 31, wid = tid >> 5;
    const int sign = wid & 1, tset = wid >> 1;
    const int laneHi = (lane >> 4) & 1;
    const int laneLo = lane & 15;

    // --- weight-producer role: this thread always fills elements (m0,k),(m1,k)
    const int kk = tid & 31;
    const int m0 = tid >> 5, m1 = m0 + 8;
    const int r0 = rowbase + m0, r1 = rowbase + m1;
    const float pR0 = p[r0],              pR1 = p[r1];
    const float mp0 = stats[0 * NV + r0], mp1 = stats[0 * NV + r1];
    const float izp0 = stats[1 * NV + r0], izp1 = stats[1 * NV + r1];
    const float mn0 = stats[2 * NV + r0], mn1 = stats[2 * NV + r1];
    const float izn0 = stats[3 * NV + r0], izn1 = stats[3 * NV + r1];
    const unsigned* mrow0 = mask + (size_t)r0 * 128;
    const unsigned* mrow1 = mask + (size_t)r1 * 128;

    auto make_weights = [&](int chunk, int buf) {
        const float qv = q[chunk * 32 + kk];
        {
            float sel = (float)((mrow0[chunk] >> kk) & 1u);
            float s = lrelu(pR0 + qv);
            float ep = __expf(fminf(s - mp0, 60.f));
            float en = __expf(fminf(-s - mn0, 60.f));
            Wlds[buf][0][m0 * 32 + kk] = (__bf16)(sel * ep * izp0);
            Wlds[buf][1][m0 * 32 + kk] = (__bf16)(sel * en * izn0);
        }
        {
            float sel = (float)((mrow1[chunk] >> kk) & 1u);
            float s = lrelu(pR1 + qv);
            float ep = __expf(fminf(s - mp1, 60.f));
            float en = __expf(fminf(-s - mn1, 60.f));
            Wlds[buf][0][m1 * 32 + kk] = (__bf16)(sel * ep * izp1);
            Wlds[buf][1][m1 * 32 + kk] = (__bf16)(sel * en * izn1);
        }
    };

    v8f acc[5] = {};

    make_weights(0, 0);
    __syncthreads();

    for (int chunk = 0; chunk < 128; ++chunk) {
        const int buf = chunk & 1;
        const int jbase = chunk * 32;

        // A fragment (16x32 bf16) from LDS, contiguous k order (k = 16*laneHi + s)
        const __bf16* ab = &Wlds[buf][sign][laneLo * 32 + laneHi * 16];
        v8bf a0 = *(const v8bf*)ab;            // K 0..7   (hi lanes: 16..23)
        v8bf a1 = *(const v8bf*)(ab + 8);      // K 8..15  (hi lanes: 24..31)
        v16bf af = __builtin_shufflevector(a0, a1, 0, 1, 2, 3, 4, 5, 6, 7,
                                           8, 9, 10, 11, 12, 13, 14, 15);
        // 5 column tiles: B from pre-transposed bf16 X (contiguous 32B/lane)
        #pragma unroll
        for (int qq = 0; qq < 5; ++qq) {
            int col = (tset * 5 + qq) * 16 + laneLo;
            const __bf16* bp = XT + (size_t)col * NV + jbase + laneHi * 16;
            v8bf b0 = *(const v8bf*)bp;
            v8bf b1 = *(const v8bf*)(bp + 8);
            v16bf bf_ = __builtin_shufflevector(b0, b1, 0, 1, 2, 3, 4, 5, 6, 7,
                                                8, 9, 10, 11, 12, 13, 14, 15);
            acc[qq] = __builtin_amdgcn_wmma_f32_16x16x32_bf16(
                false, af, false, bf_, (short)0, acc[qq], false, false);
        }

        if (chunk + 1 < 128) make_weights(chunk + 1, buf ^ 1);
        __syncthreads();
    }

    float* outp = sign ? aggN : aggP;
    #pragma unroll
    for (int qq = 0; qq < 5; ++qq) {
        int col = (tset * 5 + qq) * 16 + laneLo;
        int rbase = rowbase + laneHi * 8;           // C/D layout: hi lanes M+8
        #pragma unroll
        for (int r = 0; r < 8; ++r)
            outp[(size_t)(rbase + r) * PITCH + col] = acc[qq][r];
    }
}

// ---- x = elu( [aggP, -aggN] @ wtrans1 )  -> [4096][320] f32 (zero padded)
__global__ void k_out1(const float* aggP, const float* aggN, const float* Wt, float* x) {
    int c = blockIdx.x * 16 + threadIdx.x;  // 0..319
    int i = blockIdx.y * 16 + threadIdx.y;  // 0..4095
    if (c >= DM) { x[(size_t)i * PITCH + c] = 0.f; return; }
    float acc = 0.f;
    for (int d = 0; d < DM; ++d) {
        acc += aggP[(size_t)i * PITCH + d] * Wt[d * DM + c];
        acc -= aggN[(size_t)i * PITCH + d] * Wt[(DM + d) * DM + c];
    }
    x[(size_t)i * PITCH + c] = (acc > 0.f) ? acc : (__expf(acc) - 1.f);
}

// ---- final: out = sigmoid( [aggP2, -aggN2] @ wtrans_o )[X_tid]   (64 x 300)
__global__ void k_final(const float* aggP, const float* aggN, const float* Wt,
                        const long long* xtid, float* out, int nout) {
    int t = blockIdx.x * blockDim.x + threadIdx.x;
    if (t >= nout) return;
    int ti = t / DM, c = t % DM;
    long long i = xtid[ti];
    float acc = 0.f;
    for (int d = 0; d < DM; ++d) {
        acc += aggP[(size_t)i * PITCH + d] * Wt[d * DM + c];
        acc -= aggN[(size_t)i * PITCH + d] * Wt[(DM + d) * DM + c];
    }
    out[t] = 1.f / (1.f + __expf(-acc));
}

// ---------------------------------------------------------------------------
// workspace layout (bytes)
constexpr size_t OFF_VEC  = 0;                                 // 6*304 f32
constexpr size_t OFF_F    = 8192;                              // f,g,p1,q1,p2,q2
constexpr size_t OFF_T    = OFF_F    + 6ull * NV * 4;          // thr
constexpr size_t OFF_ST1  = OFF_T    + (size_t)NV * 4;         // 4*NV f32
constexpr size_t OFF_ST2  = OFF_ST1  + 4ull * NV * 4;
constexpr size_t OFF_MASK = OFF_ST2  + 4ull * NV * 4;          // NV*128 u32
constexpr size_t OFF_XT1  = OFF_MASK + (size_t)NV * 128 * 4;   // 320*NV bf16
constexpr size_t OFF_XT2  = OFF_XT1  + (size_t)PITCH * NV * 2;
constexpr size_t OFF_X    = OFF_XT2  + (size_t)PITCH * NV * 2; // NV*320 f32
constexpr size_t OFF_AP1  = OFF_X    + (size_t)NV * PITCH * 4;
constexpr size_t OFF_AN1  = OFF_AP1  + (size_t)NV * PITCH * 4;
constexpr size_t OFF_AP2  = OFF_AN1  + (size_t)NV * PITCH * 4;
constexpr size_t OFF_AN2  = OFF_AP2  + (size_t)NV * PITCH * 4;

extern "C" void kernel_launch(void* const* d_in, const int* in_sizes, int n_in,
                              void* d_out, int out_size, void* d_ws, size_t ws_size,
                              hipStream_t stream) {
    (void)in_sizes; (void)n_in; (void)ws_size;
    const float* emb = (const float*)d_in[0];
    const float* Wd  = (const float*)d_in[1];
    const float* a1  = (const float*)d_in[2];
    const float* a2  = (const float*)d_in[3];
    const float* W1  = (const float*)d_in[4];
    const float* ah  = (const float*)d_in[5];
    const float* wt1 = (const float*)d_in[6];
    const float* Wo  = (const float*)d_in[7];
    const float* ao  = (const float*)d_in[8];
    const float* wto = (const float*)d_in[9];
    const int*   adj = (const int*)d_in[10];
    const long long* xtid = (const long long*)d_in[11];
    float* out = (float*)d_out;

    char* ws = (char*)d_ws;
    float*    vecs  = (float*)(ws + OFF_VEC);
    float*    f     = (float*)(ws + OFF_F);
    float*    g     = f + NV;
    float*    p1    = f + 2 * NV;
    float*    q1    = f + 3 * NV;
    float*    p2    = f + 4 * NV;
    float*    q2    = f + 5 * NV;
    float*    thr   = (float*)(ws + OFF_T);
    float*    st1   = (float*)(ws + OFF_ST1);
    float*    st2   = (float*)(ws + OFF_ST2);
    unsigned* mask  = (unsigned*)(ws + OFF_MASK);
    __bf16*   XT1   = (__bf16*)(ws + OFF_XT1);
    __bf16*   XT2   = (__bf16*)(ws + OFF_XT2);
    float*    x     = (float*)(ws + OFF_X);
    float*    aggP1 = (float*)(ws + OFF_AP1);
    float*    aggN1 = (float*)(ws + OFF_AN1);
    float*    aggP2 = (float*)(ws + OFF_AP2);
    float*    aggN2 = (float*)(ws + OFF_AN2);

    // stage 0: tiny matvecs + projections + bf16 transpose + S threshold
    k_smallvec<<<8, 256, 0, stream>>>(Wd, a1, a2, W1, ah, Wo, ao, vecs);
    k_proj2<<<NV / 4, 128, 0, stream>>>(emb, DM, vecs + 0 * 304, vecs + 1 * 304, f, g);
    k_proj2<<<NV / 4, 128, 0, stream>>>(emb, DM, vecs + 2 * 304, vecs + 3 * 304, p1, q1);
    k_transpose_bf<<<dim3(NV / 256, PITCH), 256, 0, stream>>>(emb, DM, XT1);
    k_sthresh<<<NV / 4, 128, 0, stream>>>(f, g, thr);

    // layer 1
    k_mask_stats<<<NV / 4, 128, 0, stream>>>(f, g, thr, adj, p1, q1, mask, st1, 1);
    k_agg_wmma<<<NV / 16, 256, 0, stream>>>(XT1, p1, q1, st1, mask, aggP1, aggN1);
    k_out1<<<dim3(PITCH / 16, NV / 16), dim3(16, 16), 0, stream>>>(aggP1, aggN1, wt1, x);

    // layer 2
    k_transpose_bf<<<dim3(NV / 256, PITCH), 256, 0, stream>>>(x, PITCH, XT2);
    k_proj2<<<NV / 4, 128, 0, stream>>>(x, PITCH, vecs + 4 * 304, vecs + 5 * 304, p2, q2);
    k_mask_stats<<<NV / 4, 128, 0, stream>>>(f, g, thr, adj, p2, q2, mask, st2, 0);
    k_agg_wmma<<<NV / 16, 256, 0, stream>>>(XT2, p2, q2, st2, mask, aggP2, aggN2);

    // gather + sigmoid
    k_final<<<(out_size + 255) / 256, 256, 0, stream>>>(aggP2, aggN2, wto, xtid, out, out_size);
}
